// EDA_CS_65773129171177
// MI455X (gfx1250) — matrix operand order repaired and checked by reading, not compiled, not run on hardware
//
#include <hip/hip_runtime.h>
#include <hip/hip_bf16.h>

// ---------------------------------------------------------------------------
// Pointer-generator seq2seq forward for MI455X (gfx1250), wave32 + WMMA bf16.
// B=64, S=512, T=512, H=512, E=64, A=256, V=128, I=2H+E=1088, G=3H=1536.
// All GEMMs run on v_wmma_f32_16x16x32_bf16 with fp32 accumulation.
// ---------------------------------------------------------------------------

#define BB   64
#define SS   512
#define TT   512
#define HH   512
#define EE   64
#define AA   256
#define VV   128
#define GG   1536
#define SOS_TOK 1

typedef __bf16 bf16;
typedef __attribute__((ext_vector_type(16))) __bf16 v16bf;
typedef __attribute__((ext_vector_type(8)))  float  v8f;

// ---- bf16 <-> f32 helpers (RNE-ish) ---------------------------------------
__device__ __forceinline__ bf16 f2bf(float f) {
  union { float f; unsigned u; } v; v.f = f;
  unsigned r = v.u + 0x7fffu + ((v.u >> 16) & 1u);
  unsigned short h = (unsigned short)(r >> 16);
  bf16 b; __builtin_memcpy(&b, &h, 2); return b;
}
__device__ __forceinline__ float bf2f(bf16 b) {
  unsigned short h; __builtin_memcpy(&h, &b, 2);
  union { unsigned u; float f; } v; v.u = ((unsigned)h) << 16; return v.f;
}

// ---- WMMA fragment loaders (CDNA5 wave32 layouts, CDNA5 ISA) ----
// A 16x32 bf16: lanes 0-15 row m0+l hold K {k0+0..7, k0+16..23};
//               lanes 16-31 row m0+(l-16) hold K {k0+8..15, k0+24..31}.
__device__ __forceinline__ v16bf load_a_frag(const bf16* A, int lda,
                                             int m0, int k0, int lane) {
  int r = m0 + (lane & 15);
  const bf16* p = A + (long long)r * lda + k0 + ((lane >> 4) << 3);
  v16bf a;
#pragma unroll
  for (int i = 0; i < 8; ++i) { a[i] = p[i]; a[i + 8] = p[16 + i]; }
  return a;
}
// Same A layout but gathering from a per-lane fp32 row (embedding row).
__device__ __forceinline__ v16bf load_a_frag_f32row(const float* rowp,
                                                    int k0, int lane) {
  const float* p = rowp + k0 + ((lane >> 4) << 3);
  v16bf a;
#pragma unroll
  for (int i = 0; i < 8; ++i) { a[i] = f2bf(p[i]); a[i + 8] = f2bf(p[16 + i]); }
  return a;
}
// B 32x16 bf16 with B[k][n] = W[n][k], W row-major (N,K):
// lanes 0-15 col n0+l hold K {k0+0..7, k0+16..23}; lanes 16-31 hold K {k0+8..15, k0+24..31}.
__device__ __forceinline__ v16bf load_b_frag(const bf16* W, int ldw,
                                             int n0, int k0, int lane) {
  int n = n0 + (lane & 15);
  const bf16* p = W + (long long)n * ldw + k0 + ((lane >> 4) << 3);
  v16bf b;
#pragma unroll
  for (int i = 0; i < 8; ++i) { b[i] = p[i]; b[i + 8] = p[16 + i]; }
  return b;
}
__device__ __forceinline__ v8f wmma_bf16(v16bf a, v16bf b, v8f c) {
  return __builtin_amdgcn_wmma_f32_16x16x32_bf16(false, a, false, b,
                                                 (short)0, c, false, false);
}
// C/D 16x16 f32: VGPR r -> row m0 + r + 8*(lane>=16), col n0 + (lane&15).
__device__ __forceinline__ void store_c(float* dst, long long ldd,
                                        v8f c, int m0, int n0, int lane) {
  int col = n0 + (lane & 15);
  int rbase = m0 + ((lane >> 4) << 3);
#pragma unroll
  for (int r = 0; r < 8; ++r) dst[(long long)(rbase + r) * ldd + col] = c[r];
}

__device__ __forceinline__ float sigmoidf(float x) {
  return 1.f / (1.f + expf(-x));
}

// ---------------------------------------------------------------------------
// Fused GRU step: gi = [ctx|emb(tok)] @ Wih.T, gh = h @ Whh.T, gate math.
// Grid: 32 blocks (16 h-columns each), block: 128 threads (4 waves).
// Wave w covers batch rows 16w..16w+15. K-outer / gate-inner: one A-fragment
// feeds the r/z/n gate WMMAs (6 live f32 accumulators = 48 VGPRs).
// ---------------------------------------------------------------------------
__global__ __launch_bounds__(128) void gru_step_kernel(
    const bf16* __restrict__ ctxA, int K1,               // dense input part
    const float* __restrict__ emb,
    const int* __restrict__ tokens, int tok_stride, int tok_off, int use_sos,
    const bf16* __restrict__ Wih, const float* __restrict__ bih,
    const bf16* __restrict__ Whh, const float* __restrict__ bhh,
    const bf16* __restrict__ h_in_bf, int h_in_ld,
    const float* __restrict__ h_in_f32,
    float* __restrict__ h_out_f32,
    bf16* __restrict__ h_out_bf, int h_out_ld,
    bf16* __restrict__ ann_out, long long ann_stride) {
  __shared__ float s_gi[3][BB][16];
  __shared__ float s_gh[3][BB][16];

  int lane = threadIdx.x & 31;
  int w    = threadIdx.x >> 5;
  int m0   = w * 16;
  int col0 = blockIdx.x * 16;
  int KI   = K1 + EE;

  int brow = m0 + (lane & 15);
  int tok  = use_sos ? SOS_TOK : tokens[brow * tok_stride + tok_off];
  const float* erow = emb + tok * EE;

  v8f gi[3]; v8f gh[3];
#pragma unroll
  for (int g = 0; g < 3; ++g) { gi[g] = {}; gh[g] = {}; }

  // --- input gates: embedding segment (K=64, conversion hoisted out of loop)
  {
    v16bf ea0 = load_a_frag_f32row(erow, 0, lane);
    v16bf ea1 = load_a_frag_f32row(erow, 32, lane);
#pragma unroll
    for (int g = 0; g < 3; ++g) {
      int n0 = g * HH + col0;
      v16bf b0 = load_b_frag(Wih, KI, n0, K1, lane);
      gi[g] = wmma_bf16(ea0, b0, gi[g]);
      v16bf b1 = load_b_frag(Wih, KI, n0, K1 + 32, lane);
      gi[g] = wmma_bf16(ea1, b1, gi[g]);
    }
  }
  // --- input gates: dense ctx segment (decoder only; K1 = 1024)
  for (int k0 = 0; k0 < K1; k0 += 32) {
    v16bf a = load_a_frag(ctxA, K1, m0, k0, lane);
#pragma unroll
    for (int g = 0; g < 3; ++g) {
      v16bf b = load_b_frag(Wih, KI, g * HH + col0, k0, lane);
      gi[g] = wmma_bf16(a, b, gi[g]);
    }
  }
  // --- hidden gates over K = 512
  for (int k0 = 0; k0 < HH; k0 += 32) {
    v16bf a = load_a_frag(h_in_bf, h_in_ld, m0, k0, lane);
#pragma unroll
    for (int g = 0; g < 3; ++g) {
      v16bf b = load_b_frag(Whh, HH, g * HH + col0, k0, lane);
      gh[g] = wmma_bf16(a, b, gh[g]);
    }
  }
  {
    int col = lane & 15;
    int rbase = m0 + ((lane >> 4) << 3);
#pragma unroll
    for (int g = 0; g < 3; ++g)
#pragma unroll
      for (int r = 0; r < 8; ++r) {
        s_gi[g][rbase + r][col] = gi[g][r];
        s_gh[g][rbase + r][col] = gh[g][r];
      }
  }
  __syncthreads();

  // Gate math: r,z,n -> h'
  for (int e = threadIdx.x; e < BB * 16; e += 128) {
    int b = e >> 4, c = e & 15;
    int j = col0 + c;
    float gir = s_gi[0][b][c] + bih[j];
    float ghr = s_gh[0][b][c] + bhh[j];
    float giz = s_gi[1][b][c] + bih[HH + j];
    float ghz = s_gh[1][b][c] + bhh[HH + j];
    float gin = s_gi[2][b][c] + bih[2 * HH + j];
    float ghn = s_gh[2][b][c] + bhh[2 * HH + j];
    float r = sigmoidf(gir + ghr);
    float z = sigmoidf(giz + ghz);
    float n = tanhf(gin + r * ghn);
    float hold = h_in_f32[b * HH + j];
    float h2 = (1.f - z) * n + z * hold;
    h_out_f32[b * HH + j] = h2;
    bf16 hb = f2bf(h2);
    h_out_bf[(long long)b * h_out_ld + j] = hb;
    if (ann_out) ann_out[(long long)b * ann_stride + j] = hb;
  }
}

// ---------------------------------------------------------------------------
// Generic 2-segment-A WMMA GEMM: C = A @ W.T, A bf16 (M,K), W bf16 (N,K).
// grid.x = m-tile; wave handles NT consecutive n-tiles so one A-fragment
// feeds NT WMMAs. Block = 256 threads (8 waves).
// ---------------------------------------------------------------------------
template <int NT>
__global__ __launch_bounds__(256) void gemm_kernel(
    const bf16* __restrict__ A0, int lda0, int K1,
    const bf16* __restrict__ A1, int lda1,
    const bf16* __restrict__ W, int ldw,
    float* __restrict__ C, long long ldc, int Ktot) {
  int lane  = threadIdx.x & 31;
  int w     = threadIdx.x >> 5;
  int m0    = blockIdx.x * 16;
  int nbase = (blockIdx.y * 8 + w) * (NT * 16);
  v8f acc[NT];
#pragma unroll
  for (int j = 0; j < NT; ++j) acc[j] = {};
  for (int k0 = 0; k0 < Ktot; k0 += 32) {
    v16bf a = (k0 < K1) ? load_a_frag(A0, lda0, m0, k0, lane)
                        : load_a_frag(A1, lda1, m0, k0 - K1, lane);
#pragma unroll
    for (int j = 0; j < NT; ++j) {
      v16bf b = load_b_frag(W, ldw, nbase + j * 16, k0, lane);
      acc[j] = wmma_bf16(a, b, acc[j]);
    }
  }
#pragma unroll
  for (int j = 0; j < NT; ++j) store_c(C, ldc, acc[j], m0, nbase + j * 16, lane);
}

// ---------------------------------------------------------------------------
// scores[b,s] = sum_a va[a]*tanh(WaAnn[b,s,a] + qU[b,a]); masked by lengths.
// One wave per (b,s). grid = B*S/8 blocks of 256.
// ---------------------------------------------------------------------------
__global__ __launch_bounds__(256) void scores_kernel(
    const float* __restrict__ WaAnn, const float* __restrict__ qU,
    const float* __restrict__ va, const int* __restrict__ lengths,
    float* __restrict__ scores) {
  int lane = threadIdx.x & 31;
  int w    = threadIdx.x >> 5;
  int idx  = blockIdx.x * 8 + w;          // b*S + s
  int b = idx >> 9, s = idx & (SS - 1);
  const float* wa = WaAnn + (long long)idx * AA;
  const float* qb = qU + b * AA;
  float sum = 0.f;
#pragma unroll
  for (int i = 0; i < AA / 32; ++i) {
    int a = lane + i * 32;
    sum += va[a] * tanhf(wa[a] + qb[a]);
  }
  for (int off = 16; off > 0; off >>= 1) sum += __shfl_down(sum, off, 32);
  if (lane == 0) scores[idx] = (s < lengths[b]) ? sum : -1e9f;
}

// softmax over S per batch row. 64 blocks x 256.
__global__ __launch_bounds__(256) void softmax_kernel(
    const float* __restrict__ scores, float* __restrict__ att) {
  __shared__ float red[256];
  int b = blockIdx.x, tid = threadIdx.x;
  float m = fmaxf(scores[b * SS + tid], scores[b * SS + 256 + tid]);
  red[tid] = m; __syncthreads();
  for (int o = 128; o > 0; o >>= 1) {
    if (tid < o) red[tid] = fmaxf(red[tid], red[tid + o]);
    __syncthreads();
  }
  m = red[0]; __syncthreads();
  float e0 = expf(scores[b * SS + tid] - m);
  float e1 = expf(scores[b * SS + 256 + tid] - m);
  red[tid] = e0 + e1; __syncthreads();
  for (int o = 128; o > 0; o >>= 1) {
    if (tid < o) red[tid] += red[tid + o];
    __syncthreads();
  }
  float inv = 1.f / red[0];
  att[b * SS + tid] = e0 * inv;
  att[b * SS + 256 + tid] = e1 * inv;
}

// ctx[b,:] = att[b,:] @ ann[b,:,:]   (batched matvec, VALU). 64 blocks x 256.
__global__ __launch_bounds__(256) void ctx_kernel(
    const float* __restrict__ att, const bf16* __restrict__ ann,
    float* __restrict__ ctx, bf16* __restrict__ ctx_bf) {
  __shared__ float a_sh[SS];
  int b = blockIdx.x, tid = threadIdx.x;
  a_sh[tid] = att[b * SS + tid];
  a_sh[tid + 256] = att[b * SS + 256 + tid];
  __syncthreads();
  const bf16* base = ann + (long long)b * SS * (2 * HH);
  float acc0 = 0.f, acc1 = 0.f, acc2 = 0.f, acc3 = 0.f;
  for (int s = 0; s < SS; ++s) {
    float a = a_sh[s];
    const bf16* row = base + (long long)s * (2 * HH);
    acc0 += a * bf2f(row[tid]);
    acc1 += a * bf2f(row[tid + 256]);
    acc2 += a * bf2f(row[tid + 512]);
    acc3 += a * bf2f(row[tid + 768]);
  }
  ctx[b * 1024 + tid]       = acc0; ctx_bf[b * 1024 + tid]       = f2bf(acc0);
  ctx[b * 1024 + tid + 256] = acc1; ctx_bf[b * 1024 + tid + 256] = f2bf(acc1);
  ctx[b * 1024 + tid + 512] = acc2; ctx_bf[b * 1024 + tid + 512] = f2bf(acc2);
  ctx[b * 1024 + tid + 768] = acc3; ctx_bf[b * 1024 + tid + 768] = f2bf(acc3);
}

// vocab softmax + p_gen + copy distribution + mix. 64 blocks x 128.
__global__ __launch_bounds__(128) void finalize_kernel(
    const float* __restrict__ logits, const float* __restrict__ bo,
    const float* __restrict__ ctx, const float* __restrict__ hdf,
    const float* __restrict__ hdb, const float* __restrict__ emb,
    const int* __restrict__ y, int tok_off, int use_sos,
    const float* __restrict__ wg, const float* __restrict__ bg,
    const int* __restrict__ x, const float* __restrict__ att,
    float* __restrict__ out, int t) {
  __shared__ float red[VV];
  __shared__ float cp[VV];
  int b = blockIdx.x, tid = threadIdx.x;
  float l = logits[b * VV + tid] + bo[tid];
  red[tid] = l; __syncthreads();
  for (int o = 64; o > 0; o >>= 1) {
    if (tid < o) red[tid] = fmaxf(red[tid], red[tid + o]);
    __syncthreads();
  }
  float mx = red[0]; __syncthreads();
  float ev = expf(l - mx);
  red[tid] = ev; __syncthreads();
  for (int o = 64; o > 0; o >>= 1) {
    if (tid < o) red[tid] += red[tid + o];
    __syncthreads();
  }
  float pv = ev / red[0];
  __syncthreads();
  // p_gen = sigmoid([ctx, dec_out, e] . wg + bg)
  int tok = use_sos ? SOS_TOK : y[b * TT + tok_off];
  float s = 0.f;
  for (int i = tid; i < 4 * HH + EE; i += 128) {
    float v;
    if (i < 1024)      v = ctx[b * 1024 + i];
    else if (i < 1536) v = hdf[b * HH + i - 1024];
    else if (i < 2048) v = hdb[b * HH + i - 1536];
    else               v = emb[tok * EE + i - 2048];
    s += v * wg[i];
  }
  red[tid] = s; __syncthreads();
  for (int o = 64; o > 0; o >>= 1) {
    if (tid < o) red[tid] += red[tid + o];
    __syncthreads();
  }
  float pg = sigmoidf(red[0] + bg[0]);
  __syncthreads();
  cp[tid] = 0.f; __syncthreads();
  for (int si = tid; si < SS; si += 128)
    atomicAdd(&cp[x[b * SS + si]], att[b * SS + si]);
  __syncthreads();
  out[((long long)b * TT + t) * VV + tid] = pg * pv + (1.f - pg) * cp[tid];
}

// ---- small utility kernels -------------------------------------------------
__global__ void convert_kernel(bf16* __restrict__ dst,
                               const float* __restrict__ src,
                               int src_ld, int cols, long long n) {
  long long i = (long long)blockIdx.x * 256 + threadIdx.x;
  if (i < n) {
    long long r = i / cols, c = i - r * cols;
    dst[i] = f2bf(src[r * src_ld + c]);
  }
}
__global__ void enc_h_init_kernel(float* hf, float* hb, bf16* hfb, bf16* hbb) {
  int i = blockIdx.x * 256 + threadIdx.x;   // 64*512 elems
  hf[i] = 0.f; hb[i] = 0.f;
  bf16 z = f2bf(0.f);
  hfb[i] = z; hbb[i] = z;
}
__global__ void init_dec_kernel(const float* hf, const float* hb,
                                bf16* q, float* hdf, float* hdb) {
  int i = blockIdx.x * 256 + threadIdx.x;   // 64*512 elems
  int b = i >> 9, j = i & (HH - 1);
  float vf = hf[i], vb = hb[i];
  hdf[i] = vf; hdb[i] = vb;
  q[b * 1024 + j]       = f2bf(vf);
  q[b * 1024 + 512 + j] = f2bf(vb);
}
__global__ void out0_kernel(float* out) {  // out[:,0,:] = one_hot(SOS)
  int i = blockIdx.x * 256 + threadIdx.x;  // B*V
  int b = i / VV, v = i - b * VV;
  out[(long long)b * TT * VV + v] = (v == SOS_TOK) ? 1.f : 0.f;
}

// ---------------------------------------------------------------------------
extern "C" void kernel_launch(void* const* d_in, const int* in_sizes, int n_in,
                              void* d_out, int out_size, void* d_ws,
                              size_t ws_size, hipStream_t stream) {
  (void)in_sizes; (void)n_in; (void)out_size; (void)ws_size;
  const int*   x        = (const int*)d_in[0];
  const int*   lengths  = (const int*)d_in[1];
  const int*   y        = (const int*)d_in[2];
  const float* emb      = (const float*)d_in[3];
  const float* eWih[2]  = {(const float*)d_in[4],  (const float*)d_in[8]};
  const float* eWhh[2]  = {(const float*)d_in[5],  (const float*)d_in[9]};
  const float* ebih[2]  = {(const float*)d_in[6],  (const float*)d_in[10]};
  const float* ebhh[2]  = {(const float*)d_in[7],  (const float*)d_in[11]};
  const float* dWih[2]  = {(const float*)d_in[12], (const float*)d_in[16]};
  const float* dWhh[2]  = {(const float*)d_in[13], (const float*)d_in[17]};
  const float* dbih[2]  = {(const float*)d_in[14], (const float*)d_in[18]};
  const float* dbhh[2]  = {(const float*)d_in[15], (const float*)d_in[19]};
  const float* Wa = (const float*)d_in[20];
  const float* Ua = (const float*)d_in[21];
  const float* va = (const float*)d_in[22];
  const float* Wo = (const float*)d_in[23];
  const float* bo = (const float*)d_in[24];
  const float* wg = (const float*)d_in[25];
  const float* bg = (const float*)d_in[26];
  float* out = (float*)d_out;

  // ---- workspace carve ----
  char* wp = (char*)d_ws;
  auto alloc = [&](size_t bytes) -> void* {
    void* r = (void*)wp;
    wp += (bytes + 255) & ~(size_t)255;
    return r;
  };
  bf16* w_eWih[2]; bf16* w_eWhh[2]; bf16* w_dWih[2]; bf16* w_dWhh[2];
  for (int d = 0; d < 2; ++d) {
    w_eWih[d] = (bf16*)alloc((size_t)GG * EE * 2);          // sliced cols 0..63
    w_eWhh[d] = (bf16*)alloc((size_t)GG * HH * 2);
    w_dWih[d] = (bf16*)alloc((size_t)GG * (2 * HH + EE) * 2);
    w_dWhh[d] = (bf16*)alloc((size_t)GG * HH * 2);
  }
  bf16* w_Wa = (bf16*)alloc((size_t)AA * 2 * HH * 2);
  bf16* w_Ua = (bf16*)alloc((size_t)AA * 2 * HH * 2);
  bf16* w_Wo = (bf16*)alloc((size_t)VV * 4 * HH * 2);
  bf16* ann  = (bf16*)alloc((size_t)BB * SS * 2 * HH * 2);  // 67 MB
  float* WaAnn = (float*)alloc((size_t)BB * SS * AA * 4);   // 33.5 MB
  float* hef_f[2]; float* hef_b[2]; bf16* hefb_f[2]; bf16* hefb_b[2];
  float* hdf[2]; float* hdb[2]; bf16* qb[2];
  for (int p = 0; p < 2; ++p) {
    hef_f[p]  = (float*)alloc((size_t)BB * HH * 4);
    hef_b[p]  = (float*)alloc((size_t)BB * HH * 4);
    hefb_f[p] = (bf16*)alloc((size_t)BB * HH * 2);
    hefb_b[p] = (bf16*)alloc((size_t)BB * HH * 2);
    hdf[p]    = (float*)alloc((size_t)BB * HH * 4);
    hdb[p]    = (float*)alloc((size_t)BB * HH * 4);
    qb[p]     = (bf16*)alloc((size_t)BB * 2 * HH * 2);
  }
  float* qU     = (float*)alloc((size_t)BB * AA * 4);
  float* scores = (float*)alloc((size_t)BB * SS * 4);
  float* att    = (float*)alloc((size_t)BB * SS * 4);
  float* ctx    = (float*)alloc((size_t)BB * 2 * HH * 4);
  bf16*  ctx_bf = (bf16*)alloc((size_t)BB * 2 * HH * 2);
  float* logits = (float*)alloc((size_t)BB * VV * 4);

  auto conv = [&](bf16* dst, const float* src, int rows, int src_ld, int cols) {
    long long n = (long long)rows * cols;
    convert_kernel<<<dim3((unsigned)((n + 255) / 256)), dim3(256), 0, stream>>>(
        dst, src, src_ld, cols, n);
  };
  // ---- weight conversion (encoder Wih sliced to its live 64 columns) ----
  for (int d = 0; d < 2; ++d) {
    conv(w_eWih[d], eWih[d], GG, 2 * HH + EE, EE);
    conv(w_eWhh[d], eWhh[d], GG, HH, HH);
    conv(w_dWih[d], dWih[d], GG, 2 * HH + EE, 2 * HH + EE);
    conv(w_dWhh[d], dWhh[d], GG, HH, HH);
  }
  conv(w_Wa, Wa, AA, 2 * HH, 2 * HH);
  conv(w_Ua, Ua, AA, 2 * HH, 2 * HH);
  conv(w_Wo, Wo, VV, 4 * HH, 4 * HH);

  // ---- init ----
  enc_h_init_kernel<<<dim3(BB * HH / 256), dim3(256), 0, stream>>>(
      hef_f[0], hef_b[0], hefb_f[0], hefb_b[0]);
  out0_kernel<<<dim3(BB * VV / 256), dim3(256), 0, stream>>>(out);

  // ---- encoder: 512 steps, forward + backward, ping-pong hidden state ----
  int pp = 0;
  for (int t = 0; t < SS; ++t) {
    gru_step_kernel<<<dim3(HH / 16), dim3(128), 0, stream>>>(
        nullptr, 0, emb, x, SS, t, 0,
        w_eWih[0], ebih[0], w_eWhh[0], ebhh[0],
        hefb_f[pp], HH, hef_f[pp],
        hef_f[1 - pp], hefb_f[1 - pp], HH,
        ann + (long long)t * (2 * HH), (long long)SS * 2 * HH);
    int tb = SS - 1 - t;
    gru_step_kernel<<<dim3(HH / 16), dim3(128), 0, stream>>>(
        nullptr, 0, emb, x, SS, tb, 0,
        w_eWih[1], ebih[1], w_eWhh[1], ebhh[1],
        hefb_b[pp], HH, hef_b[pp],
        hef_b[1 - pp], hefb_b[1 - pp], HH,
        ann + (long long)tb * (2 * HH) + HH, (long long)SS * 2 * HH);
    pp ^= 1;
  }
  // 512 flips -> final encoder states live in buffer 0.

  // ---- attention key precompute: WaAnn = ann @ Wa.T (M=32768,N=256,K=1024)
  gemm_kernel<2><<<dim3(BB * SS / 16, 1), dim3(256), 0, stream>>>(
      ann, 2 * HH, 2 * HH, nullptr, 0, w_Wa, 2 * HH, WaAnn, AA, 2 * HH);

  // ---- decoder init: q0 = [hf, hb] ----
  init_dec_kernel<<<dim3(BB * HH / 256), dim3(256), 0, stream>>>(
      hef_f[0], hef_b[0], qb[0], hdf[0], hdb[0]);

  // ---- decoder: 511 steps ----
  int pd = 0, qc = 0;
  for (int i = 0; i < TT - 1; ++i) {
    int use_sos = (i == 0) ? 1 : 0;
    // qU = q @ Ua.T  (M=64, N=256, K=1024)
    gemm_kernel<2><<<dim3(BB / 16, 1), dim3(256), 0, stream>>>(
        qb[qc], 2 * HH, 2 * HH, nullptr, 0, w_Ua, 2 * HH, qU, AA, 2 * HH);
    scores_kernel<<<dim3(BB * SS / 8), dim3(256), 0, stream>>>(
        WaAnn, qU, va, lengths, scores);
    softmax_kernel<<<dim3(BB), dim3(256), 0, stream>>>(scores, att);
    ctx_kernel<<<dim3(BB), dim3(256), 0, stream>>>(att, ann, ctx, ctx_bf);
    // decoder GRU cells: gin = [ctx (1024) | emb(tok) (64)]
    gru_step_kernel<<<dim3(HH / 16), dim3(128), 0, stream>>>(
        ctx_bf, 2 * HH, emb, y, TT, i, use_sos,
        w_dWih[0], dbih[0], w_dWhh[0], dbhh[0],
        qb[qc], 2 * HH, hdf[pd],
        hdf[1 - pd], qb[1 - qc], 2 * HH,
        nullptr, 0);
    gru_step_kernel<<<dim3(HH / 16), dim3(128), 0, stream>>>(
        ctx_bf, 2 * HH, emb, y, TT, i, use_sos,
        w_dWih[1], dbih[1], w_dWhh[1], dbhh[1],
        qb[qc] + HH, 2 * HH, hdb[pd],
        hdb[1 - pd], qb[1 - qc] + HH, 2 * HH,
        nullptr, 0);
    // logits = [dec_out | ctx] @ Wo.T  (M=64, N=128, K=2048)
    gemm_kernel<1><<<dim3(BB / 16, 1), dim3(256), 0, stream>>>(
        qb[1 - qc], 2 * HH, 2 * HH, ctx_bf, 2 * HH, w_Wo, 4 * HH,
        logits, VV, 4 * HH);
    finalize_kernel<<<dim3(BB), dim3(128), 0, stream>>>(
        logits, bo, ctx, hdf[1 - pd], hdb[1 - pd], emb, y, i, use_sos,
        wg, bg, x, att, out, i + 1);
    pd ^= 1; qc ^= 1;
  }
}
